// GRU2_16904991277109
// MI455X (gfx1250) — matrix operand order repaired and checked by reading, not compiled, not run on hardware
//
#include <hip/hip_runtime.h>
#include <hip/hip_bf16.h>

// ---------------- problem constants ----------------
#define T_SEQ 1024
#define BATCH 32
#define DIN   512
#define HID   512
#define G3    1536          // 3*H

// ---------------- types ----------------
typedef __bf16 bf16_t;
typedef bf16_t v16bf __attribute__((ext_vector_type(16)));
typedef bf16_t v8bf  __attribute__((ext_vector_type(8)));
typedef float  v8f   __attribute__((ext_vector_type(8)));
typedef float  v4f   __attribute__((ext_vector_type(4)));
typedef unsigned v2u __attribute__((ext_vector_type(2)));

static __device__ __forceinline__ unsigned short f2bf_raw(float f) {
  unsigned u = __builtin_bit_cast(unsigned, f);
  return (unsigned short)((u + 0x7FFFu + ((u >> 16) & 1u)) >> 16);  // RNE
}
static __device__ __forceinline__ bf16_t f2bf(float f) {
  return __builtin_bit_cast(bf16_t, f2bf_raw(f));
}
static __device__ __forceinline__ unsigned pk2(float a, float b) {
  return (unsigned)f2bf_raw(a) | ((unsigned)f2bf_raw(b) << 16);
}
static __device__ __forceinline__ v2u pk4(v4f v) {
  v2u r; r.x = pk2(v.x, v.y); r.y = pk2(v.z, v.w); return r;
}
static __device__ __forceinline__ v16bf cat16(v8bf lo, v8bf hi) {
  return __builtin_shufflevector(lo, hi, 0,1,2,3,4,5,6,7,8,9,10,11,12,13,14,15);
}
static __device__ __forceinline__ float sigf(float x) {
  return 1.0f / (1.0f + __expf(-x));
}

// =====================================================================
// Kernel 1: input projection GEMM (both directions)
//   ih[dir][t*B+b][g] = sum_d x[b,t,d]*w_ih[g,d] + b_ih[g] + (g<2H ? b_hh[g] : 0)
// Block tile 128(M) x 128(N), 8 waves (2x4), each wave 64x32 (4x2 WMMA frags),
// K staged 64-deep: vectorized b128 global loads -> packed bf16 -> ds_store_b64.
// =====================================================================
#define ASTR 72   // padded K stride (bf16) for 64-deep K chunk; 144B = 9*16

__global__ __launch_bounds__(256) void gru_ih_gemm(
    const float* __restrict__ x,
    const float* __restrict__ wih_fw, const float* __restrict__ bih_fw, const float* __restrict__ bhh_fw,
    const float* __restrict__ wih_bw, const float* __restrict__ bih_bw, const float* __restrict__ bhh_bw,
    float* __restrict__ ih_ws)
{
  __shared__ __align__(16) bf16_t lds_a[128 * ASTR];   // 128 rows x 64 k
  __shared__ __align__(16) bf16_t lds_b[128 * ASTR];   // 128 w-rows x 64 k

  const int dir = blockIdx.z;
  const float* wih = dir ? wih_bw : wih_fw;
  const float* bih = dir ? bih_bw : bih_fw;
  const float* bhh = dir ? bhh_bw : bhh_fw;
  float* ih = ih_ws + (size_t)dir * (size_t)T_SEQ * BATCH * G3;

  const int m0 = blockIdx.x * 128;
  const int n0 = blockIdx.y * 128;
  const int tid  = threadIdx.x;
  const int wave = tid >> 5, lane = tid & 31;
  const int wm = wave >> 2, wn = wave & 3;     // 2 x 4 wave grid
  const int lane_n = lane & 15, hig = lane >> 4;

  v8f acc[4][2] = {};

  for (int kc = 0; kc < DIN; kc += 64) {
    __syncthreads();
    // stage A: 128 rows x 64 k  (2048 float4; 8 per thread), batched loads
    {
      v4f ta[8];
#pragma unroll
      for (int i = 0; i < 8; ++i) {
        int v = tid + i * 256;
        int row = v >> 4, kv = (v & 15) * 4;
        int mg = m0 + row;
        int t = mg >> 5, b = mg & 31;
        ta[i] = *(const v4f*)&x[((size_t)(b * T_SEQ + t)) * DIN + kc + kv];
      }
#pragma unroll
      for (int i = 0; i < 8; ++i) {
        int v = tid + i * 256;
        int row = v >> 4, kv = (v & 15) * 4;
        *(v2u*)&lds_a[row * ASTR + kv] = pk4(ta[i]);
      }
    }
    // stage B: 128 w-rows x 64 k
    {
      v4f tb[8];
#pragma unroll
      for (int i = 0; i < 8; ++i) {
        int v = tid + i * 256;
        int row = v >> 4, kv = (v & 15) * 4;
        tb[i] = *(const v4f*)&wih[(size_t)(n0 + row) * DIN + kc + kv];
      }
#pragma unroll
      for (int i = 0; i < 8; ++i) {
        int v = tid + i * 256;
        int row = v >> 4, kv = (v & 15) * 4;
        *(v2u*)&lds_b[row * ASTR + kv] = pk4(tb[i]);
      }
    }
    __syncthreads();

#pragma unroll
    for (int ks = 0; ks < 64; ks += 32) {
      v16bf afr[4], bfr[2];
#pragma unroll
      for (int i = 0; i < 4; ++i) {
        // A frag (16x32): lane<16 -> K 0..7 & 16..23 ; lane>=16 -> K 8..15 & 24..31
        int row = wm * 64 + i * 16 + lane_n;
        const bf16_t* p = &lds_a[row * ASTR + ks + hig * 8];
        afr[i] = cat16(*(const v8bf*)p, *(const v8bf*)(p + 16));
      }
#pragma unroll
      for (int j = 0; j < 2; ++j) {
        // B frag (32x16): lane<16 -> (N=lane, K 0..7 & 16..23); lane>=16 -> (N=lane-16, K 8..15 & 24..31)
        int col = wn * 32 + j * 16 + lane_n;
        const bf16_t* q = &lds_b[col * ASTR + ks + hig * 8];
        bfr[j] = cat16(*(const v8bf*)q, *(const v8bf*)(q + 16));
      }
#pragma unroll
      for (int i = 0; i < 4; ++i)
#pragma unroll
        for (int j = 0; j < 2; ++j)
          acc[i][j] = __builtin_amdgcn_wmma_f32_16x16x32_bf16(
              false, afr[i], false, bfr[j], (short)0, acc[i][j], false, false);
    }
  }

  // epilogue: D frag element r -> M = r + 8*(lane>>4), N = lane&15
#pragma unroll
  for (int i = 0; i < 4; ++i) {
#pragma unroll
    for (int j = 0; j < 2; ++j) {
      int n = n0 + wn * 32 + j * 16 + lane_n;
      float bias = bih[n] + (n < 2 * HID ? bhh[n] : 0.0f);
#pragma unroll
      for (int r = 0; r < 8; ++r) {
        int m = m0 + wm * 64 + i * 16 + r + 8 * hig;
        ih[(size_t)m * G3 + n] = acc[i][j][r] + bias;
      }
    }
  }
}

// =====================================================================
// Kernel 2: recurrence. 16 blocks: 0..7 = fw, 8..15 = bw (concurrent).
// Each block keeps its w_hh slice (3 gates x 64 cols x 512 K, bf16, ~195 KB)
// resident in LDS for all 1024 steps. h state double-buffered in global (L2);
// one atomics grid barrier per step. 12 waves = (gate x 4 ntiles);
// 2 M-frags x 16 K-chunks -> 32 WMMA / wave / step.
// =====================================================================
#define RBLK     8
#define RCOLS    64
#define RTHREADS 384
#define WSTRIDE  520   // padded K stride (bf16); 1040B = 130*8 (b64-aligned)

__global__ __launch_bounds__(RTHREADS) void gru_recurrent(
    const float* __restrict__ whh_fw, const float* __restrict__ bhh_fw,
    const float* __restrict__ whh_bw, const float* __restrict__ bhh_bw,
    const float* __restrict__ ih_ws,
    float* __restrict__ h_ws,         // [dir][2][B][H]
    unsigned* __restrict__ sync_ws,   // per-dir {cnt, gen} in separate lines
    float* __restrict__ out,          // (B, T, 2H)
    float* __restrict__ hlast)        // [dir][B][H]
{
  extern __shared__ __align__(16) char smem[];
  bf16_t* lds_w  = (bf16_t*)smem;                      // 192 x WSTRIDE
  bf16_t* lds_h  = lds_w + 192 * WSTRIDE;              // 32  x WSTRIDE
  float*  lds_hh = (float*)(lds_h + 32 * WSTRIDE);     // 3 x 32 x 64

  const int dir  = blockIdx.x / RBLK;
  const int jb   = blockIdx.x % RBLK;
  const int col0 = jb * RCOLS;
  const float* whh = dir ? whh_bw : whh_fw;
  const float* bhh = dir ? bhh_bw : bhh_fw;
  const float* ih  = ih_ws + (size_t)dir * T_SEQ * BATCH * G3;
  float* hbuf = h_ws + (size_t)dir * 2 * BATCH * HID;
  unsigned* cnt = sync_ws + dir * 64;
  unsigned* gen = sync_ws + dir * 64 + 32;

  const int tid  = threadIdx.x;
  const int wave = tid >> 5, lane = tid & 31;
  const int gate = wave >> 2, nt = wave & 3;
  const int lane_n = lane & 15, hig = lane >> 4;

  // resident w_hh slice -> LDS bf16 (24576 float4 total; 64 per thread)
#pragma unroll 8
  for (int i = 0; i < 64; ++i) {
    int v = tid + i * RTHREADS;
    int lr = v >> 7, kv = (v & 127) * 4;
    int grow = (lr >> 6) * HID + col0 + (lr & 63);
    v4f w = *(const v4f*)&whh[(size_t)grow * HID + kv];
    *(v2u*)&lds_w[lr * WSTRIDE + kv] = pk4(w);
  }

  for (int tt = 0; tt < T_SEQ; ++tt) {
    const int t = dir ? (T_SEQ - 1 - tt) : tt;
    const float* hr = hbuf + (tt & 1) * BATCH * HID;
    float*       hw = hbuf + ((tt + 1) & 1) * BATCH * HID;

    // ---- grid barrier across the 8 blocks of this direction ----
    __threadfence();            // make prev-step h writes agent-visible
    __syncthreads();
    if (tid == 0) {
      unsigned g = __hip_atomic_load(gen, __ATOMIC_RELAXED, __HIP_MEMORY_SCOPE_AGENT);
      unsigned a = __hip_atomic_fetch_add(cnt, 1u, __ATOMIC_ACQ_REL, __HIP_MEMORY_SCOPE_AGENT);
      if (a == RBLK - 1) {
        __hip_atomic_store(cnt, 0u, __ATOMIC_RELAXED, __HIP_MEMORY_SCOPE_AGENT);
        __hip_atomic_fetch_add(gen, 1u, __ATOMIC_RELEASE, __HIP_MEMORY_SCOPE_AGENT);
      } else {
        while (__hip_atomic_load(gen, __ATOMIC_ACQUIRE, __HIP_MEMORY_SCOPE_AGENT) == g)
          __builtin_amdgcn_s_sleep(1);
      }
    }
    __syncthreads();

    // ---- prefetch this step's ih slice (3 gates x 32 batch x 64 cols) ----
    if (tid < 192) {
      int half = tid & 1, seg = tid >> 1;          // 96 row-segments x 2 lines
      int gp = seg >> 5, m = seg & 31;
      __builtin_prefetch(&ih[((size_t)t * BATCH + m) * G3 + gp * HID + col0 + half * 32], 0, 3);
    }

    // ---- stage h(prev) into LDS as bf16 (4096 float4) ----
    for (int v = tid; v < (BATCH * HID) / 4; v += RTHREADS) {
      int m = v >> 7, kv = (v & 127) * 4;
      v4f h4 = *(const v4f*)&hr[m * HID + kv];
      *(v2u*)&lds_h[m * WSTRIDE + kv] = pk4(h4);
    }
    __syncthreads();

    // ---- hh = h @ w_hh_slice^T (per-wave 32x16 output, f32 accum) ----
    v8f acc0 = {}, acc1 = {};
    const int brow = gate * RCOLS + nt * 16 + lane_n;
#pragma unroll
    for (int kc = 0; kc < HID; kc += 32) {
      const bf16_t* q = &lds_w[brow * WSTRIDE + kc + hig * 8];
      v16bf bfr = cat16(*(const v8bf*)q, *(const v8bf*)(q + 16));
      const bf16_t* pa0 = &lds_h[lane_n * WSTRIDE + kc + hig * 8];
      v16bf a0 = cat16(*(const v8bf*)pa0, *(const v8bf*)(pa0 + 16));
      const bf16_t* pa1 = &lds_h[(16 + lane_n) * WSTRIDE + kc + hig * 8];
      v16bf a1 = cat16(*(const v8bf*)pa1, *(const v8bf*)(pa1 + 16));
      acc0 = __builtin_amdgcn_wmma_f32_16x16x32_bf16(false, a0, false, bfr, (short)0, acc0, false, false);
      acc1 = __builtin_amdgcn_wmma_f32_16x16x32_bf16(false, a1, false, bfr, (short)0, acc1, false, false);
    }
    {
      int c = nt * 16 + lane_n;
#pragma unroll
      for (int r = 0; r < 8; ++r) {
        lds_hh[gate * (BATCH * RCOLS) + (r + 8 * hig) * RCOLS + c]      = acc0[r];
        lds_hh[gate * (BATCH * RCOLS) + (16 + r + 8 * hig) * RCOLS + c] = acc1[r];
      }
    }
    __syncthreads();

    // ---- gates + state update + output (vectorized x4) ----
    for (int v = tid; v < (BATCH * RCOLS) / 4; v += RTHREADS) {
      int m = v >> 4, c4 = (v & 15) * 4;
      int cg = col0 + c4;
      size_t base = ((size_t)t * BATCH + m) * G3;
      v4f ir = *(const v4f*)&ih[base + cg];
      v4f iz = *(const v4f*)&ih[base + HID + cg];
      v4f in = *(const v4f*)&ih[base + 2 * HID + cg];
      v4f hhr = *(const v4f*)&lds_hh[m * RCOLS + c4];
      v4f hhz = *(const v4f*)&lds_hh[BATCH * RCOLS + m * RCOLS + c4];
      v4f hhn = *(const v4f*)&lds_hh[2 * BATCH * RCOLS + m * RCOLS + c4];
      v4f bn  = *(const v4f*)&bhh[2 * HID + cg];
      v4f hp  = *(const v4f*)&hr[m * HID + cg];
      v4f hv;
#pragma unroll
      for (int e = 0; e < 4; ++e) {
        float rg = sigf(ir[e] + hhr[e]);
        float zg = sigf(iz[e] + hhz[e]);
        float ng = tanhf(in[e] + rg * (hhn[e] + bn[e]));   // b_hh_n inside r*( )
        hv[e] = (1.0f - zg) * ng + zg * hp[e];
      }
      *(v4f*)&hw[m * HID + cg] = hv;
      *(v4f*)&out[((size_t)m * T_SEQ + t) * (2 * HID) + dir * HID + cg] = hv;
      if (tt == T_SEQ - 1) *(v4f*)&hlast[(size_t)dir * BATCH * HID + m * HID + cg] = hv;
    }
  }
}

// =====================================================================
// Kernel 0: init h state + barrier counters (ws is poisoned, graph replays)
// =====================================================================
__global__ void gru_init(float* __restrict__ h_ws, unsigned* __restrict__ sync_ws) {
  int i = blockIdx.x * blockDim.x + threadIdx.x;
  if (i < 2 * 2 * BATCH * HID) h_ws[i] = 0.0f;
  if (i < 128) sync_ws[i] = 0u;
}

// =====================================================================
extern "C" void kernel_launch(void* const* d_in, const int* in_sizes, int n_in,
                              void* d_out, int out_size, void* d_ws, size_t ws_size,
                              hipStream_t stream) {
  (void)in_sizes; (void)n_in; (void)out_size; (void)ws_size;
  const float* x      = (const float*)d_in[0];
  const float* wih_fw = (const float*)d_in[1];
  const float* whh_fw = (const float*)d_in[2];
  const float* bih_fw = (const float*)d_in[3];
  const float* bhh_fw = (const float*)d_in[4];
  const float* wih_bw = (const float*)d_in[5];
  const float* whh_bw = (const float*)d_in[6];
  const float* bih_bw = (const float*)d_in[7];
  const float* bhh_bw = (const float*)d_in[8];

  float* out   = (float*)d_out;
  float* hlast = out + (size_t)BATCH * T_SEQ * 2 * HID;

  // workspace layout
  const size_t IH_ELEMS = (size_t)2 * T_SEQ * BATCH * G3;    // ~403 MB f32
  float*    ih_ws   = (float*)d_ws;
  float*    h_ws    = ih_ws + IH_ELEMS;                      // 2*2*B*H
  unsigned* sync_ws = (unsigned*)(h_ws + 2 * 2 * BATCH * HID);

  gru_init<<<(2 * 2 * BATCH * HID + 255) / 256, 256, 0, stream>>>(h_ws, sync_ws);

  dim3 g1(32768 / 128, G3 / 128, 2);
  gru_ih_gemm<<<g1, 256, 0, stream>>>(x, wih_fw, bih_fw, bhh_fw,
                                      wih_bw, bih_bw, bhh_bw, ih_ws);

  const int smem = (192 * WSTRIDE + 32 * WSTRIDE) * (int)sizeof(bf16_t)
                 + 3 * BATCH * RCOLS * (int)sizeof(float);   // 257,536 B < 320 KB
  (void)hipFuncSetAttribute((const void*)gru_recurrent,
                            hipFuncAttributeMaxDynamicSharedMemorySize, smem);
  gru_recurrent<<<2 * RBLK, RTHREADS, smem, stream>>>(
      whh_fw, bhh_fw, whh_bw, bhh_bw, ih_ws, h_ws, sync_ws, out, hlast);
}
